// SparseNet_618475290897
// MI455X (gfx1250) — compile-verified, matrix operand order checked by reading, not run on hardware
//
#include <hip/hip_runtime.h>
#include <hip/hip_bf16.h>

// ---------------- problem constants (B=1, C=256, H=92, W=124) ----------------
#define CDEPTH 256
#define HH     92
#define WW     124
#define NPIX   (HH * WW)           // 11408 = 713 * 16 exactly
#define NT     (NPIX / 16)         // 713 tiles of 16 along both n and m
#define KSTEPS (CDEPTH / 32)       // 8 wmma k-steps of 32
#define TILE_US (KSTEPS * 32 * 16) // 4096 ushorts = 8KB per packed 16-wide tile
#define CSPLIT 8                   // column-range split (machine fill)
#define KNN    4
#define SR     4
#define WIN    9
#define CELLS  (WIN * WIN)         // 81
#define LEVELS 5
#define PPB    128                 // pixels per block in volume kernel

typedef __attribute__((ext_vector_type(16))) __bf16 v16bf;
typedef __attribute__((ext_vector_type(8)))  float  v8f;

#if defined(__has_builtin)
#  if __has_builtin(__builtin_amdgcn_sched_group_barrier)
#    define SGB(mask, size, id) __builtin_amdgcn_sched_group_barrier(mask, size, id)
#  endif
#endif
#ifndef SGB
#  define SGB(mask, size, id)
#endif

// branchless sorted-top4 bubble insert (desc order in tv0..tv3)
#define TOP4_INS(v, id)                                                        \
    {                                                                          \
        bool c = (v) > tv3;                                                    \
        tv3 = c ? (v) : tv3;  ti3 = c ? (id) : ti3;                            \
        c = tv3 > tv2; { float tf = tv2; int tj = ti2;                         \
            tv2 = c ? tv3 : tv2; ti2 = c ? ti3 : ti2;                          \
            tv3 = c ? tf  : tv3; ti3 = c ? tj  : ti3; }                        \
        c = tv2 > tv1; { float tf = tv1; int tj = ti1;                         \
            tv1 = c ? tv2 : tv1; ti1 = c ? ti2 : ti1;                          \
            tv2 = c ? tf  : tv2; ti2 = c ? tj  : ti2; }                        \
        c = tv1 > tv0; { float tf = tv0; int tj = ti0;                         \
            tv0 = c ? tv1 : tv0; ti0 = c ? ti1 : ti0;                          \
            tv1 = c ? tf  : tv1; ti1 = c ? tj  : ti1; }                        \
    }

// =====================================================================
// Kernel 1: pack f32 feature maps into bf16 WMMA per-lane operand layout
//   dst layout: [tile][kstep][lane][16 bf16]  (32B per lane per k-step)
// =====================================================================
__global__ __launch_bounds__(256) void sparse_pack_kernel(
    const float* __restrict__ f1, const float* __restrict__ f2,
    unsigned short* __restrict__ apk, unsigned short* __restrict__ bpk)
{
    const int g = blockIdx.x * blockDim.x + threadIdx.x;   // one (mat,tile,kstep,lane)
    const int per_mat = NT * KSTEPS * 32;
    if (g >= 2 * per_mat) return;
    const int mat  = g / per_mat;
    int rem        = g - mat * per_mat;
    const int tile = rem / (KSTEPS * 32);
    rem           -= tile * (KSTEPS * 32);
    const int s    = rem >> 5;
    const int lane = rem & 31;
    const int lr   = lane & 15;
    const int hi   = lane >> 4;
    const int col  = tile * 16 + lr;          // n for A-pack, m for B-pack

    const float* __restrict__ src = mat ? f2 : f1;
    unsigned short* __restrict__ dst =
        (mat ? bpk : apk) + (((size_t)tile * KSTEPS + s) * 32 + lane) * 16;

    unsigned short vals[16];
#pragma unroll
    for (int j = 0; j < 16; ++j) {
        const int k = (j < 8) ? (hi * 8 + j) : (16 + hi * 8 + (j - 8));
        const int c = s * 32 + k;
        unsigned int u = __float_as_uint(src[(size_t)c * NPIX + col]);
        u += 0x7fffu + ((u >> 16) & 1u);      // round-to-nearest-even bf16
        vals[j] = (unsigned short)(u >> 16);
    }
#pragma unroll
    for (int j = 0; j < 16; ++j) dst[j] = vals[j];  // merged -> b128 stores
}

// =====================================================================
// Kernel 2: scores GEMM (bf16 WMMA) + per-row running top-4 over a
//   column slice. Grid = (90 row-groups) x (CSPLIT column slices).
//   Wave w owns row tile rt = blk.x*8 + w; the 8KB B tile is staged in
//   LDS once per workgroup (double-buffered). Slice-local top-4 goes to
//   a candidate buffer merged by a follow-up kernel.
// =====================================================================
__global__ __launch_bounds__(256, 1) void sparse_score_topk_kernel(
    const unsigned short* __restrict__ apk,
    const unsigned short* __restrict__ bpk,
    float* __restrict__ candV, int* __restrict__ candI)
{
    __shared__ uint4 sB[2][TILE_US / 8];     // 2 x 8KB double-buffered B tile
    __shared__ float s_score[8][256];        // per-wave 16x16 C staging
    __shared__ float s_cv[8][16][8];         // per-wave merge candidates
    __shared__ int   s_ci[8][16][8];

    const int tid  = threadIdx.x;
    const int wave = tid >> 5;
    const int lane = tid & 31;
    const int lr   = lane & 15;
    const int hi   = lane >> 4;
    const int rt_raw = blockIdx.x * 8 + wave;
    const bool rt_valid = (rt_raw < NT);
    const int rt   = rt_valid ? rt_raw : (NT - 1); // duplicate work, identical results
    const int ys   = blockIdx.y;                   // column slice
    const int t_lo = (ys * NT) / CSPLIT;
    const int t_hi = ((ys + 1) * NT) / CSPLIT;

    // ---- A operand regs for this wave's row tile (64 VGPRs) ----
    v16bf a[KSTEPS];
#pragma unroll
    for (int s = 0; s < KSTEPS; ++s) {
        const unsigned short* ap = apk + (((size_t)rt * KSTEPS + s) * 32 + lane) * 16;
        a[s] = *(const v16bf*)ap;
    }

    // ---- prime the pipeline: stage first slice tile into buffer 0 ----
    {
        const uint4* gsrc = (const uint4*)(bpk + (size_t)t_lo * TILE_US);
        sB[0][tid * 2 + 0] = gsrc[tid * 2 + 0];
        sB[0][tid * 2 + 1] = gsrc[tid * 2 + 1];
    }
    __syncthreads();

    // per-lane partial top-4: lane handles row=lr, cols [hi*8, hi*8+8)
    float tv0 = -1e30f, tv1 = -1e30f, tv2 = -1e30f, tv3 = -1e30f;
    int   ti0 = 0, ti1 = 0, ti2 = 0, ti3 = 0;

    for (int t = t_lo; t < t_hi; ++t) {
        const int cur = (t - t_lo) & 1;
        const bool have_next = (t + 1 < t_hi);

        // (1) issue next tile's global loads early (overlap with compute)
        uint4 n0, n1;
        if (have_next) {
            const uint4* gsrc = (const uint4*)(bpk + (size_t)(t + 1) * TILE_US);
            n0 = gsrc[tid * 2 + 0];
            n1 = gsrc[tid * 2 + 1];
            if (t + 2 < t_hi)
                __builtin_prefetch(bpk + (size_t)(t + 2) * TILE_US + tid * 16, 0, 1);
        }

        // (2) pull all 8 B k-step operands from LDS, then wait-free WMMA chain
        const v16bf* bl = (const v16bf*)&sB[cur][0];
        v16bf b[KSTEPS];
#pragma unroll
        for (int s = 0; s < KSTEPS; ++s) b[s] = bl[s * 32 + lane];
        v8f acc = {};
#pragma unroll
        for (int s = 0; s < KSTEPS; ++s)
            acc = __builtin_amdgcn_wmma_f32_16x16x32_bf16(
                      false, a[s], false, b[s], (short)0, acc, false, false);

        // (3) stage C tile per-wave and scan: lane v holds score[v+8*hi][lr]
#pragma unroll
        for (int v = 0; v < 8; ++v)
            s_score[wave][(v + 8 * hi) * 16 + lr] = acc[v];
        // same-wave DS ops are in-order: read back without a barrier
        const float4* sp = (const float4*)&s_score[wave][lr * 16 + hi * 8];
        const float4 q0 = sp[0], q1 = sp[1];

        // schedule shaping: globals first, then 16 DS reads, then 8 WMMAs
        // back-to-back (forces distinct B regs -> one dscnt wait per tile),
        // then the C staging writes and scan reads.
        SGB(0x010, 3, 0);    // VMEM: n0, n1, prefetch
        SGB(0x100, 16, 0);   // 16 ds_load_b128 (B operands)
        SGB(0x008, 8, 0);    // 8 v_wmma back-to-back
        SGB(0x200, 4, 0);    // staging ds writes
        SGB(0x100, 2, 0);    // scan ds reads

        const float m = fmaxf(fmaxf(fmaxf(q0.x, q0.y), fmaxf(q0.z, q0.w)),
                              fmaxf(fmaxf(q1.x, q1.y), fmaxf(q1.z, q1.w)));
        if (m > tv3) {                      // cheap guard; inserts are branchless
            const int idb = t * 16 + hi * 8;
            TOP4_INS(q0.x, idb + 0); TOP4_INS(q0.y, idb + 1);
            TOP4_INS(q0.z, idb + 2); TOP4_INS(q0.w, idb + 3);
            TOP4_INS(q1.x, idb + 4); TOP4_INS(q1.y, idb + 5);
            TOP4_INS(q1.z, idb + 6); TOP4_INS(q1.w, idb + 7);
        }

        // (4) commit next tile into the other LDS buffer, then rendezvous
        if (have_next) {
            sB[cur ^ 1][tid * 2 + 0] = n0;
            sB[cur ^ 1][tid * 2 + 1] = n1;
        }
        __syncthreads();
    }

    // ---- wave-local merge: row lr has 2 partial top-4s (lane halves) ----
    s_cv[wave][lr][hi * 4 + 0] = tv0;  s_ci[wave][lr][hi * 4 + 0] = ti0;
    s_cv[wave][lr][hi * 4 + 1] = tv1;  s_ci[wave][lr][hi * 4 + 1] = ti1;
    s_cv[wave][lr][hi * 4 + 2] = tv2;  s_ci[wave][lr][hi * 4 + 2] = ti2;
    s_cv[wave][lr][hi * 4 + 3] = tv3;  s_ci[wave][lr][hi * 4 + 3] = ti3;
    // in-wave DS ordering; lanes 0..15 merge the 8 candidates of their row
    if (lane < 16 && rt_valid) {
        float tv0 = -1e30f, tv1 = -1e30f, tv2 = -1e30f, tv3 = -1e30f;
        int   ti0 = 0, ti1 = 0, ti2 = 0, ti3 = 0;
#pragma unroll
        for (int c = 0; c < 8; ++c) {
            const float v = s_cv[wave][lane][c];
            const int  id = s_ci[wave][lane][c];
            TOP4_INS(v, id);
        }
        const int n = rt * 16 + lane;
        const size_t base = ((size_t)n * CSPLIT + ys) * 4;
        candV[base + 0] = tv0;  candI[base + 0] = ti0;
        candV[base + 1] = tv1;  candI[base + 1] = ti1;
        candV[base + 2] = tv2;  candI[base + 2] = ti2;
        candV[base + 3] = tv3;  candI[base + 3] = ti3;
    }
}

// =====================================================================
// Kernel 2b: merge CSPLIT slice-local top-4 lists -> final idx + corr.
// =====================================================================
__global__ __launch_bounds__(256) void sparse_merge_kernel(
    const float* __restrict__ candV, const int* __restrict__ candI,
    int* __restrict__ idx_out, float* __restrict__ corr_out)
{
    const int n = blockIdx.x * blockDim.x + threadIdx.x;
    if (n >= NPIX) return;
    float tv0 = -1e30f, tv1 = -1e30f, tv2 = -1e30f, tv3 = -1e30f;
    int   ti0 = 0, ti1 = 0, ti2 = 0, ti3 = 0;
    const size_t base = (size_t)n * CSPLIT * 4;
#pragma unroll
    for (int c = 0; c < CSPLIT * 4; ++c) {
        const float v = candV[base + c];
        const int  id = candI[base + c];
        TOP4_INS(v, id);
    }
    idx_out[n * 4 + 0] = ti0;  corr_out[n * 4 + 0] = tv0 * 0.0625f; // /sqrt(256)
    idx_out[n * 4 + 1] = ti1;  corr_out[n * 4 + 1] = tv1 * 0.0625f;
    idx_out[n * 4 + 2] = ti2;  corr_out[n * 4 + 2] = tv2 * 0.0625f;
    idx_out[n * 4 + 3] = ti3;  corr_out[n * 4 + 3] = tv3 * 0.0625f;
}

// =====================================================================
// Kernel 3: 5-level bilinear scatter into dense [405, H, W] volume.
//   Scatter is pixel-local -> thread-private 81-float LDS slab, no atomics.
// =====================================================================
__global__ __launch_bounds__(PPB, 1) void sparse_volume_kernel(
    const float* __restrict__ dflow, const int* __restrict__ idx,
    const float* __restrict__ corr, float* __restrict__ out)
{
    __shared__ float acc[PPB * CELLS];  // 128 * 81 * 4B = 41.5 KB
    const int t = threadIdx.x;
    const int n = blockIdx.x * PPB + t;
    const bool ok = (n < NPIX);

    float ccy[KNN], ccx[KNN], cr[KNN];
    if (ok) {
        const int y0 = n / WW, x0 = n - y0 * WW;
        const float dfx = dflow[n];            // delta_flow channel 0 (x)
        const float dfy = dflow[NPIX + n];     // delta_flow channel 1 (y)
#pragma unroll
        for (int k = 0; k < KNN; ++k) {
            const int m  = idx[n * KNN + k];
            const int my = m / WW, mx = m - my * WW;
            ccy[k] = (float)(my - y0) - dfy;   // cc = disp - delta_flow[:, [1,0]]
            ccx[k] = (float)(mx - x0) - dfx;
            cr[k]  = corr[n * KNN + k];
        }
    }

    float scale = 1.0f;
    for (int lev = 0; lev < LEVELS; ++lev) {
#pragma unroll
        for (int c = 0; c < CELLS; ++c) acc[t * CELLS + c] = 0.0f;
        if (ok) {
#pragma unroll
            for (int k = 0; k < KNN; ++k) {
                const float csy = ccy[k] * scale, csx = ccx[k] * scale;
                const float yf = floorf(csy), xf = floorf(csx);
                const float dy = csy - yf,   dx = csx - xf;
#pragma unroll
                for (int c = 0; c < 4; ++c) {           // 4 bilinear corners
                    const float cy = yf + (float)(c >> 1);
                    const float cx = xf + (float)(c & 1);
                    const float w  = ((c >> 1) ? dy : 1.0f - dy) *
                                     ((c & 1)  ? dx : 1.0f - dx);
                    if (fabsf(cy) <= (float)SR && fabsf(cx) <= (float)SR) {
                        int iy = (int)cy + SR; int ix = (int)cx + SR;
                        iy = min(max(iy, 0), WIN - 1);
                        ix = min(max(ix, 0), WIN - 1);
                        acc[t * CELLS + iy * WIN + ix] += w * cr[k];
                    }
                }
            }
            // channel-major coalesced write-out; stride 81 -> no bank conflicts
            for (int ch = 0; ch < CELLS; ++ch)
                out[((size_t)(lev * CELLS + ch)) * NPIX + n] = acc[t * CELLS + ch];
        }
        scale *= 0.5f;
    }
}

// =====================================================================
extern "C" void kernel_launch(void* const* d_in, const int* in_sizes, int n_in,
                              void* d_out, int out_size, void* d_ws, size_t ws_size,
                              hipStream_t stream) {
    (void)in_sizes; (void)n_in; (void)out_size; (void)ws_size;
    const float* fmap1 = (const float*)d_in[0];   // [256, 92, 124] f32
    const float* fmap2 = (const float*)d_in[1];
    const float* dflow = (const float*)d_in[2];   // [2, 92, 124] f32
    float* out = (float*)d_out;                   // [405, 92, 124] f32

    const size_t packElems = (size_t)NT * TILE_US;          // per matrix, bf16
    unsigned short* apk = (unsigned short*)d_ws;
    unsigned short* bpk = apk + packElems;
    float* candV  = (float*)(bpk + packElems);
    int*   candI  = (int*)(candV + (size_t)NPIX * CSPLIT * 4);
    int*   idxbuf = (int*)(candI + (size_t)NPIX * CSPLIT * 4);
    float* corrbuf = (float*)(idxbuf + (size_t)NPIX * KNN);

    {   // pack both operands into WMMA per-lane bf16 layout
        const int total  = 2 * NT * KSTEPS * 32;
        const int blocks = (total + 255) / 256;
        sparse_pack_kernel<<<blocks, 256, 0, stream>>>(fmap1, fmap2, apk, bpk);
    }
    {   // bf16 WMMA scores + slice-local top-4 (90 row-groups x 8 col slices)
        dim3 grid((NT + 7) / 8, CSPLIT);
        sparse_score_topk_kernel<<<grid, 256, 0, stream>>>(apk, bpk, candV, candI);
    }
    {   // merge slice candidates -> final knn idx + corr
        const int blocks = (NPIX + 255) / 256;
        sparse_merge_kernel<<<blocks, 256, 0, stream>>>(candV, candI, idxbuf, corrbuf);
    }
    {   // dense 405-channel cost volume
        const int blocks = (NPIX + PPB - 1) / PPB;
        sparse_volume_kernel<<<blocks, PPB, 0, stream>>>(dflow, idxbuf, corrbuf, out);
    }
}
